// RelativePositionAttention_67894843015792
// MI455X (gfx1250) — matrix operand order repaired and checked by reading, not compiled, not run on hardware
//
#include <hip/hip_runtime.h>
#include <hip/hip_bf16.h>

typedef __attribute__((ext_vector_type(16))) _Float16 v16h;
typedef __attribute__((ext_vector_type(8)))  _Float16 v8h;
typedef __attribute__((ext_vector_type(8)))  float    v8f;

constexpr int   CB  = 16;    // batch
constexpr int   CN  = 577;   // tokens
constexpr int   CE  = 768;   // embed
constexpr int   CH  = 12;    // heads
constexpr int   CD  = 64;    // head dim
constexpr int   CL  = 14;    // rpe length
constexpr int   CT  = 30;    // table rows
constexpr int   CNP = 608;   // N padded to multiple of 32
constexpr int   CGS = 24;    // grid side sqrt(576)
constexpr float CSCALE = 0.125f; // 64^-0.5

static __device__ __forceinline__ v8f wmma_f16(v16h a, v16h b, v8f c) {
    return __builtin_amdgcn_wmma_f32_16x16x32_f16(false, a, false, b, (short)0, c, false, false);
}

// ---- x f32 -> f16 (one pass; keeps the GEMM inner loops cvt-free) ----
__global__ void k_xh(const float* x, _Float16* xh) {
    int gid = blockIdx.x * blockDim.x + threadIdx.x;
    if (gid < CB * CN * CE) xh[gid] = (_Float16)x[gid];
}

// ---- transpose + f16-convert the four 768x768 weights: wt[w][n][k] = W[k][n] ----
__global__ void k_wt(const float* Wq, const float* Wk, const float* Wv, const float* Wp,
                     _Float16* wt) {
    int gid = blockIdx.x * blockDim.x + threadIdx.x;
    if (gid >= 4 * CE * CE) return;
    int w = gid / (CE * CE);
    int r = gid % (CE * CE);
    int kk = r / CE, n = r % CE;
    const float* src = (w == 0) ? Wq : (w == 1) ? Wk : (w == 2) ? Wv : Wp;
    wt[(size_t)w * CE * CE + (size_t)n * CE + kk] = (_Float16)src[(size_t)kk * CE + n];
}

// ---- packed relative-position indices: idx[n][m] = iv | (ih<<8) ----
__global__ void k_idx(int* idx) {
    int gid = blockIdx.x * blockDim.x + threadIdx.x;
    if (gid >= CN * CN) return;
    int n = gid / CN, m = gid % CN;
    int iv = 0, ih = 0;
    if (n > 0 && m > 0) {
        int r = n - 1, c = m - 1;
        int dv = c / CGS - r / CGS;
        int dh = c % CGS - r % CGS;
        dv = dv < -CL ? -CL : (dv > CL ? CL : dv);
        dh = dh < -CL ? -CL : (dh > CL ? CL : dh);
        iv = dv + CL + 1;
        ih = dh + CL + 1;
    }
    idx[gid] = iv | (ih << 8);
}

// ---- QKV projection: one wave = one 16x64 C strip (a full head), A reused 4x ----
__global__ __launch_bounds__(32) void k_qkv(const _Float16* xh, const _Float16* wt,
                                            _Float16* qws, _Float16* kws, _Float16* vtws) {
    const int lane = threadIdx.x;
    const int m = lane & 15, half = lane >> 4;
    const int bt = blockIdx.x;             // 0..(16*38-1)
    const int b = bt / (CNP / 16), nt = bt % (CNP / 16);
    const int n0 = nt * 16;
    const int h  = blockIdx.y;             // head
    const int c0 = h * CD;
    const int z  = blockIdx.z;             // 0=q 1=k 2=v

    const _Float16* wz = wt + (size_t)z * CE * CE;
    const int nrow = n0 + m;
    const bool valid = nrow < CN;
    const _Float16* xrow = xh + ((size_t)b * CN + (valid ? nrow : 0)) * CE;

    v8f acc[4] = {{}, {}, {}, {}};
    for (int kc = 0; kc < CE; kc += 32) {
        v16h a;
        if (valid) {
            const v8h lo = *(const v8h*)(xrow + kc + half * 16);
            const v8h hi = *(const v8h*)(xrow + kc + half * 16 + 8);
#pragma unroll
            for (int i = 0; i < 8; ++i) { a[i] = lo[i]; a[8 + i] = hi[i]; }
        } else {
#pragma unroll
            for (int i = 0; i < 16; ++i) a[i] = (_Float16)0.f;
        }
#pragma unroll
        for (int ct = 0; ct < 4; ++ct) {
            const v16h bf = *(const v16h*)(wz + (size_t)(c0 + ct * 16 + m) * CE + kc + half * 16);
            acc[ct] = wmma_f16(a, bf, acc[ct]);
        }
    }

    if (z < 2) {
        _Float16* dst = (z == 0 ? qws : kws) + (size_t)(b * CH + h) * CNP * CD;
        const float sc = (z == 0) ? CSCALE : 1.0f;
#pragma unroll
        for (int ct = 0; ct < 4; ++ct)
#pragma unroll
            for (int v = 0; v < 8; ++v) {
                const int row = v + 8 * half;
                dst[(size_t)(n0 + row) * CD + ct * 16 + m] = (_Float16)(acc[ct][v] * sc);
            }
    } else {
#pragma unroll
        for (int ct = 0; ct < 4; ++ct) {
            _Float16* dst = vtws + ((size_t)(b * CH + h) * CD + ct * 16 + m) * CNP
                                 + n0 + 8 * half;
            v8h pk;
#pragma unroll
            for (int v = 0; v < 8; ++v) pk[v] = (_Float16)acc[ct][v];
            *(v8h*)dst = pk;
        }
    }
}

// ---- fused attention: async K staging -> S=QK^T + RPE, softmax, buckets, O=PV ----
__global__ __launch_bounds__(128) void k_attn(
    const _Float16* q, const _Float16* k, const _Float16* vt, const int* idx,
    const float* tvk, const float* thk, const float* tvv, const float* thv,
    _Float16* aout)
{
    __shared__ float S[16][612];
    __shared__ float QTV[16][32];
    __shared__ float QTH[16][32];
    __shared__ float BV[16][32];
    __shared__ float BH[16][32];
    __shared__ float red[16][8];
    __shared__ float rowmax[16];
    __shared__ float rowsum[16];
    __shared__ __attribute__((aligned(64))) _Float16 KTILE[4][2][16 * CD]; // per-wave dbl buf

    const int tid  = threadIdx.x;
    const int wave = tid >> 5, lane = tid & 31;
    const int m = lane & 15, half = lane >> 4;
    const int n0 = blockIdx.x * 16;
    const int bh = blockIdx.y;
    const int b = bh / CH, h = bh % CH;

    const _Float16* qb = q  + (size_t)bh * CNP * CD;
    const _Float16* kb = k  + (size_t)bh * CNP * CD;
    const _Float16* vb = vt + (size_t)bh * CD * CNP;

    // Q A-fragments for kc=0 and kc=32 (row m, contiguous k = 16*half + i)
    v16h a0, a1;
    {
        const _Float16* qr = qb + (size_t)(n0 + m) * CD;
        const v8h q00 = *(const v8h*)(qr + half * 16);
        const v8h q01 = *(const v8h*)(qr + half * 16 + 8);
        const v8h q10 = *(const v8h*)(qr + 32 + half * 16);
        const v8h q11 = *(const v8h*)(qr + 40 + half * 16);
#pragma unroll
        for (int i = 0; i < 8; ++i) {
            a0[i] = q00[i]; a0[8 + i] = q01[i];
            a1[i] = q10[i]; a1[8 + i] = q11[i];
        }
    }

    // zero buckets
    for (int i = tid; i < 16 * 32; i += 128) { (&BV[0][0])[i] = 0.f; (&BH[0][0])[i] = 0.f; }

    // qtv = Q @ tvk^T, qth = Q @ thk^T  (wave0 / wave1, WMMA, tables f32->f16)
    if (wave < 2) {
        const float* tab = (wave == 0) ? tvk : thk;
        float (*outp)[32] = (wave == 0) ? QTV : QTH;
#pragma unroll
        for (int t0 = 0; t0 < 32; t0 += 16) {
            v8f c = {};
            const int t = t0 + m;
#pragma unroll
            for (int kc = 0; kc < CD; kc += 32) {
                v16h bf;
                if (t < CT) {
                    const float* tr = tab + (size_t)t * CD + kc + half * 16;
#pragma unroll
                    for (int i = 0; i < 16; ++i) bf[i] = (_Float16)tr[i];
                } else {
#pragma unroll
                    for (int i = 0; i < 16; ++i) bf[i] = (_Float16)0.f;
                }
                c = wmma_f16(kc == 0 ? a0 : a1, bf, c);
            }
#pragma unroll
            for (int v = 0; v < 8; ++v) outp[v + 8 * half][t0 + m] = c[v];
        }
    }
    __syncthreads();

    // S row-block: 38 column tiles striped over 4 waves; K tiles staged into LDS
    // via async-to-LDS (ASYNCcnt) with per-wave double buffering.
    {
        const unsigned ktBase = (unsigned)(size_t)(&KTILE[wave][0][0]); // LDS byte offset
        auto issue_tile = [&](int jt, int buf) {
            // one K tile = 16 rows x 64 halfs = 2048 contiguous bytes
            const unsigned long long g =
                (unsigned long long)(size_t)((const char*)kb + (size_t)jt * 2048 + lane * 16);
            const unsigned l = ktBase + (unsigned)buf * 2048u + (unsigned)lane * 16u;
#pragma unroll
            for (int i = 0; i < 4; ++i) {
                asm volatile("global_load_async_to_lds_b128 %0, %1, off"
                             :: "v"(l + i * 512u), "v"(g + (unsigned long long)(i * 512))
                             : "memory");
            }
        };

        int ibuf = 0;
        issue_tile(wave, 0);
        for (int jt = wave; jt < CNP / 16; jt += 4, ibuf ^= 1) {
            const int jn = jt + 4;
            if (jn < CNP / 16) {
                issue_tile(jn, ibuf ^ 1);
                asm volatile("s_wait_asynccnt 0x4" ::: "memory");
            } else {
                asm volatile("s_wait_asynccnt 0x0" ::: "memory");
            }
            v8f c = {};
            const _Float16* kt = &KTILE[wave][ibuf][0] + m * CD + half * 16;
            const v16h b0 = *(const v16h*)(kt);
            const v16h b1 = *(const v16h*)(kt + 32);
            c = wmma_f16(a0, b0, c);
            c = wmma_f16(a1, b1, c);
            const int col = jt * 16 + m;
#pragma unroll
            for (int v = 0; v < 8; ++v) {
                const int row = v + 8 * half;
                const int grow = n0 + row;
                float val = -3.0e38f;
                if (col < CN && grow < CN) {
                    const int pk = idx[(size_t)grow * CN + col];
                    val = c[v] + QTV[row][pk & 255] + QTH[row][pk >> 8];
                }
                S[row][col] = val;
            }
        }
    }
    __syncthreads();

    // softmax: 8 threads per row
    {
        const int r = tid >> 3, l8 = tid & 7;
        const bool vrow = (n0 + r) < CN;
        float mx = -3.0e38f;
        for (int cc = l8; cc < CN; cc += 8) mx = fmaxf(mx, S[r][cc]);
        red[r][l8] = mx;
        __syncthreads();
        if (l8 == 0) {
            float t = red[r][0];
#pragma unroll
            for (int i = 1; i < 8; ++i) t = fmaxf(t, red[r][i]);
            rowmax[r] = t;
        }
        __syncthreads();
        const float rm = rowmax[r];
        float sm = 0.f;
        for (int cc = l8; cc < CNP; cc += 8) {
            float e = 0.f;
            if (vrow && cc < CN) e = __expf(S[r][cc] - rm);
            S[r][cc] = e;
            sm += e;
        }
        red[r][l8] = sm;
        __syncthreads();
        if (l8 == 0) {
            float t = 0.f;
#pragma unroll
            for (int i = 0; i < 8; ++i) t += red[r][i];
            rowsum[r] = fmaxf(t, 1e-20f);
        }
        __syncthreads();
        const float inv = 1.0f / rowsum[r];
        for (int cc = l8; cc < CNP; cc += 8) S[r][cc] *= inv;
    }
    __syncthreads();

    // deterministic 30-bin bucket sums of P per row (threads 0..15: V, 16..31: H)
    if (tid < 32) {
        const int r = tid & 15;
        if ((n0 + r) < CN) {
            const int* ir = idx + (size_t)(n0 + r) * CN;
            if (tid < 16) {
                for (int cc = 0; cc < CN; ++cc) BV[r][ir[cc] & 255] += S[r][cc];
            } else {
                for (int cc = 0; cc < CN; ++cc) BH[r][ir[cc] >> 8] += S[r][cc];
            }
        }
    }
    __syncthreads();

    // O = P @ V (wave w owns d-range [16w,16w+16)), plus buckets @ tvv/thv
    {
        const int d0 = wave * 16;
        v8f c = {};
        for (int kc = 0; kc < CNP; kc += 32) {
            v16h a;
            const float* pr = &S[m][0];
#pragma unroll
            for (int i = 0; i < 8; ++i) {
                a[i]     = (_Float16)pr[kc + half * 16 + i];
                a[8 + i] = (_Float16)pr[kc + half * 16 + 8 + i];
            }
            const v16h bf = *(const v16h*)(vb + (size_t)(d0 + m) * CNP + kc + half * 16);
            c = wmma_f16(a, bf, c);
        }
        const int dcol = d0 + m;
        float rpe[8];
#pragma unroll
        for (int v = 0; v < 8; ++v) rpe[v] = 0.f;
        for (int t = 0; t < CT; ++t) {
            const float av = tvv[t * CD + dcol];
            const float ah = thv[t * CD + dcol];
#pragma unroll
            for (int v = 0; v < 8; ++v)
                rpe[v] += BV[v + 8 * half][t] * av + BH[v + 8 * half][t] * ah;
        }
#pragma unroll
        for (int v = 0; v < 8; ++v) {
            const int grow = n0 + v + 8 * half;
            if (grow < CN)
                aout[((size_t)b * CN + grow) * CE + h * CD + dcol] = (_Float16)(c[v] + rpe[v]);
        }
    }
}

// ---- output projection: one wave = 16x64 C strip, out = aout @ Wp + bp ----
__global__ __launch_bounds__(32) void k_oproj(const _Float16* aout, const _Float16* wpt,
                                              const float* bp, float* out) {
    const int lane = threadIdx.x;
    const int m = lane & 15, half = lane >> 4;
    const int mt = blockIdx.x;
    const int c0 = blockIdx.y * 64;
    const _Float16* ar = aout + (size_t)(mt * 16 + m) * CE;

    v8f acc[4] = {{}, {}, {}, {}};
    for (int kc = 0; kc < CE; kc += 32) {
        v16h a;
        const v8h lo = *(const v8h*)(ar + kc + half * 16);
        const v8h hi = *(const v8h*)(ar + kc + half * 16 + 8);
#pragma unroll
        for (int i = 0; i < 8; ++i) { a[i] = lo[i]; a[8 + i] = hi[i]; }
#pragma unroll
        for (int ct = 0; ct < 4; ++ct) {
            const v16h bf = *(const v16h*)(wpt + (size_t)(c0 + ct * 16 + m) * CE + kc + half * 16);
            acc[ct] = wmma_f16(a, bf, acc[ct]);
        }
    }
#pragma unroll
    for (int ct = 0; ct < 4; ++ct) {
        const float bias = bp[c0 + ct * 16 + m];
#pragma unroll
        for (int v = 0; v < 8; ++v) {
            const int rowg = mt * 16 + v + 8 * half;
            out[(size_t)rowg * CE + c0 + ct * 16 + m] = acc[ct][v] + bias;
        }
    }
}

extern "C" void kernel_launch(void* const* d_in, const int* in_sizes, int n_in,
                              void* d_out, int out_size, void* d_ws, size_t ws_size,
                              hipStream_t stream) {
    (void)in_sizes; (void)n_in; (void)out_size; (void)ws_size;
    const float* x   = (const float*)d_in[0];
    const float* Wq  = (const float*)d_in[1];
    const float* Wk  = (const float*)d_in[2];
    const float* Wv  = (const float*)d_in[3];
    const float* Wp  = (const float*)d_in[4];
    const float* bp  = (const float*)d_in[5];
    const float* tvk = (const float*)d_in[6];
    const float* thk = (const float*)d_in[7];
    const float* tvv = (const float*)d_in[8];
    const float* thv = (const float*)d_in[9];
    float* out = (float*)d_out;

    char* ws = (char*)d_ws;
    auto al = [](size_t v) { return (v + 255) & ~(size_t)255; };
    size_t o = 0;
    _Float16* wt   = (_Float16*)(ws + o); o = al(o + (size_t)4 * CE * CE * 2);
    int*      idx  = (int*)(ws + o);      o = al(o + (size_t)CN * CN * 4);
    _Float16* xh   = (_Float16*)(ws + o); o = al(o + (size_t)CB * CN * CE * 2);
    _Float16* qws  = (_Float16*)(ws + o); o = al(o + (size_t)CB * CH * CNP * CD * 2);
    _Float16* kws  = (_Float16*)(ws + o); o = al(o + (size_t)CB * CH * CNP * CD * 2);
    _Float16* vtws = (_Float16*)(ws + o); o = al(o + (size_t)CB * CH * CD * CNP * 2);
    _Float16* aout = (_Float16*)(ws + o); o = al(o + (size_t)CB * CN * CE * 2);

    k_xh<<<dim3((CB * CN * CE + 255) / 256), dim3(256), 0, stream>>>(x, xh);
    k_wt<<<dim3((4 * CE * CE + 255) / 256), dim3(256), 0, stream>>>(Wq, Wk, Wv, Wp, wt);
    k_idx<<<dim3((CN * CN + 255) / 256), dim3(256), 0, stream>>>(idx);
    k_qkv<<<dim3(CB * (CNP / 16), CH, 3), dim3(32), 0, stream>>>(xh, wt, qws, kws, vtws);
    k_attn<<<dim3((CN + 15) / 16, CB * CH), dim3(128), 0, stream>>>(
        qws, kws, vtws, idx, tvk, thk, tvv, thv, aout);
    k_oproj<<<dim3(CB * CN / 16, CH), dim3(32), 0, stream>>>(
        aout, wt + (size_t)3 * CE * CE, bp, out);
}
